// LaneGcnHead_52776558133726
// MI455X (gfx1250) — matrix operand order repaired and checked by reading, not compiled, not run on hardware
//
#include <hip/hip_runtime.h>
#include <stdint.h>

// ---------------------------------------------------------------------------
// LaneGCN prediction head, fused for MI455X (gfx1250, wave32, WMMA).
// Every (actor, mode) row is independent (GroupNorm(1,C) is row-local), so the
// whole net is a chain of row GEMMs done with v_wmma_f32_16x16x32_bf16
// (bf16 A/B, f32 accumulate). ~48 GFLOP total, ~150MB HBM traffic -> near the
// 23.3 TB/s memory floor. v2: 2 row-tiles per wave (B-fragment reuse halves
// LDS traffic per FLOP) + global_prefetch_b8 of next-stage weights.
// ---------------------------------------------------------------------------

#define NA      32768
#define D       128
#define MODES   6
#define TPB     256
#define ROWS_WG 256          // 8 waves * 32 rows (2 tiles of 16)
#define SA      132          // LDS row stride (bf16 elems), breaks bank conflicts
#define EPSN    1e-5f
#define ACT_BLK 64           // actors per block in sort kernel

typedef __attribute__((ext_vector_type(16))) __bf16 bf16x16;
typedef __attribute__((ext_vector_type(8)))  float  f32x8;

union FragU { unsigned int u[8]; bf16x16 v; };

__device__ __forceinline__ unsigned short f2bf(float f) {
  union { float f; unsigned int u; } x; x.f = f;
  unsigned int r = x.u + 0x7FFFu + ((x.u >> 16) & 1u);   // round-nearest-even
  return (unsigned short)(r >> 16);
}
__device__ __forceinline__ float bf2f(unsigned short h) {
  union { unsigned int u; float f; } x; x.u = ((unsigned int)h) << 16;
  return x.f;
}

// A fragment (16x32 bf16): lane lo = row M; VGPR i -> K = hi*8 + (i<4 ? 2i : 16+2(i-4))
__device__ __forceinline__ bf16x16 fragA(const unsigned short* rowp, int hi, int kt) {
  FragU f; const int kb = kt * 32 + hi * 8;
#pragma unroll
  for (int i = 0; i < 8; ++i) {
    int kk = kb + ((i < 4) ? (2 * i) : (16 + 2 * (i - 4)));
    f.u[i] = *(const unsigned int*)(rowp + kk);
  }
  return f.v;
}
// B fragment (32x16 bf16, stored transposed wT[n][k]): lane lo = col N;
// VGPR i -> K = hi*8 + (i<4 ? 2i : 16+2(i-4))  (same K order as fragA)
__device__ __forceinline__ bf16x16 fragB(const unsigned short* colp, int hi, int kt) {
  FragU f; const int kb = kt * 32 + hi * 8;
#pragma unroll
  for (int i = 0; i < 8; ++i) f.u[i] = *(const unsigned int*)(colp + kb + ((i < 4) ? (2 * i) : (16 + 2 * (i - 4))));
  return f.v;
}

// Two 16x128 tiles += {A0,A1} @ 128x128 : 64 WMMAs, every B fragment used twice
__device__ __forceinline__ void mm128x2(f32x8 acc0[8], f32x8 acc1[8],
                                        const unsigned short* r0p,
                                        const unsigned short* r1p,
                                        const unsigned short* sW, int lo, int hi) {
#pragma unroll
  for (int kt = 0; kt < 4; ++kt) {
    bf16x16 fa0 = fragA(r0p, hi, kt);
    bf16x16 fa1 = fragA(r1p, hi, kt);
#pragma unroll
    for (int nt = 0; nt < 8; ++nt) {
      bf16x16 b = fragB(sW + (nt * 16 + lo) * SA, hi, kt);
      acc0[nt] = __builtin_amdgcn_wmma_f32_16x16x32_bf16(
          false, fa0, false, b, (short)0, acc0[nt], false, false);
      acc1[nt] = __builtin_amdgcn_wmma_f32_16x16x32_bf16(
          false, fa1, false, b, (short)0, acc1[nt], false, false);
    }
  }
}

// GroupNorm stats over 128 channels of 16 rows in C/D layout:
// VGPR v, lane l -> (M = v + 8*(l/16), N = nt*16 + l%16).
__device__ __forceinline__ void gn_stats(const f32x8 acc[8], float mean[8], float inv[8]) {
#pragma unroll
  for (int v = 0; v < 8; ++v) {
    float s = 0.f, q = 0.f;
#pragma unroll
    for (int nt = 0; nt < 8; ++nt) { float x = acc[nt][v]; s += x; q += x * x; }
#pragma unroll
    for (int m = 1; m < 16; m <<= 1) { s += __shfl_xor(s, m, 32); q += __shfl_xor(q, m, 32); }
    float mu  = s * (1.f / 128.f);
    float var = q * (1.f / 128.f) - mu * mu;
    mean[v] = mu; inv[v] = rsqrtf(var + EPSN);
  }
}

// GN + ReLU + bf16 store of one 16-row tile to dstRowBase (= buf + tileRow0*SA)
__device__ __forceinline__ void gn_relu_store(const f32x8 acc[8], const float* sG,
                                              const float* sB, unsigned short* dstRowBase,
                                              int lo, int hi) {
  float mean[8], inv[8];
  gn_stats(acc, mean, inv);
#pragma unroll
  for (int nt = 0; nt < 8; ++nt) {
    int col = nt * 16 + lo; float g = sG[col], b = sB[col];
#pragma unroll
    for (int v = 0; v < 8; ++v) {
      float x = (acc[nt][v] - mean[v]) * inv[v] * g + b;
      dstRowBase[(v + 8 * hi) * SA + col] = f2bf(fmaxf(x, 0.f));
    }
  }
}

struct HeadParams {
  const float *actors, *ctrs;
  const float *pW1, *pg1, *pb1, *pW2, *pg2, *pb2, *pWo, *pbo;
  const float *cW1, *cg1, *cb1, *cW2, *cg2, *cb2, *cWo, *cbo;
  const float *adW1, *adb1, *adW2, *adg2, *adb2, *aaW, *aag, *aab;
};

__device__ __forceinline__ void loadW(unsigned short* sW, const float* W, int K, int Nw, int tid) {
  for (int idx = tid; idx < K * Nw; idx += TPB) {
    int k = idx / Nw, n = idx - k * Nw;
    sW[n * SA + k] = f2bf(W[idx]);                 // coalesced read, transposed store
  }
}
__device__ __forceinline__ void loadV(float* dst, const float* src, int n, int tid) {
  for (int i = tid; i < n; i += TPB) dst[i] = src[i];
}
// global_prefetch_b8 of next stage's weights into cache while current GEMM runs
__device__ __forceinline__ void prefW(const float* p, int floats, int tid) {
  const char* c = (const char*)p;
  int bytes = floats * 4;
  for (int off = tid * 128; off < bytes; off += TPB * 128)
    __builtin_prefetch(c + off, 0, 1);
}

__global__ __launch_bounds__(TPB) void lanegcn_head_kernel(HeadParams P, float* __restrict__ reg_ws,
                                                           float* __restrict__ cls_ws) {
  __shared__ unsigned short sAct[ROWS_WG * SA];   // actors (later: cls hidden 'c')
  __shared__ unsigned short sX[ROWS_WG * SA];     // working activations
  __shared__ unsigned short sW[D * SA];           // current weights wT[n][k]
  __shared__ float sG[D], sB[D], sC[D];
  __shared__ float sDist[ROWS_WG][2];

  const int tid  = threadIdx.x;
  const int w    = tid >> 5;
  const int lane = tid & 31;
  const int lo   = lane & 15;
  const int hi   = lane >> 4;
  const int mode = blockIdx.y;
  const int r0   = blockIdx.x * ROWS_WG;
  const int rw   = r0 + w * 32;                    // this wave's first global row
  const int wr   = w * 32;                         // wave-local row base in LDS
  f32x8 acc0[8], acc1[8];
  float mean[8], inv[8];
  const f32x8 zf = {0.f, 0.f, 0.f, 0.f, 0.f, 0.f, 0.f, 0.f};

  // stage 0: actors tile -> bf16 LDS
  for (int idx = tid; idx < ROWS_WG * D; idx += TPB) {
    int r = idx >> 7, k = idx & 127;
    sAct[r * SA + k] = f2bf(P.actors[(size_t)(r0 + r) * D + k]);
  }
  __syncthreads();

  // ---- stage 1: h = relu(GN(actors @ W1[m], g1, b1)) -> sX
  loadW(sW, P.pW1 + (size_t)mode * D * D, D, D, tid);
  loadV(sG, P.pg1 + mode * D, D, tid);
  loadV(sB, P.pb1 + mode * D, D, tid);
  __syncthreads();
  prefW(P.pW2 + (size_t)mode * D * D, D * D, tid);
#pragma unroll
  for (int nt = 0; nt < 8; ++nt) { acc0[nt] = zf; acc1[nt] = zf; }
  mm128x2(acc0, acc1, sAct + (wr + lo) * SA, sAct + (wr + 16 + lo) * SA, sW, lo, hi);
  gn_relu_store(acc0, sG, sB, sX + wr * SA, lo, hi);
  gn_relu_store(acc1, sG, sB, sX + (wr + 16) * SA, lo, hi);
  __syncthreads();

  // ---- stage 2: h = relu(GN(h @ W2[m], g2, b2) + actors) -> sX
  loadW(sW, P.pW2 + (size_t)mode * D * D, D, D, tid);
  loadV(sG, P.pg2 + mode * D, D, tid);
  loadV(sB, P.pb2 + mode * D, D, tid);
  __syncthreads();
  prefW(P.pWo + (size_t)mode * D * 60, D * 60, tid);
#pragma unroll
  for (int nt = 0; nt < 8; ++nt) { acc0[nt] = zf; acc1[nt] = zf; }
  mm128x2(acc0, acc1, sX + (wr + lo) * SA, sX + (wr + 16 + lo) * SA, sW, lo, hi);
#pragma unroll
  for (int t = 0; t < 2; ++t) {
    f32x8* A = t ? acc1 : acc0;
    gn_stats(A, mean, inv);
#pragma unroll
    for (int nt = 0; nt < 8; ++nt) {
      int col = nt * 16 + lo; float g = sG[col], b = sB[col];
#pragma unroll
      for (int v = 0; v < 8; ++v) {
        float x = (A[nt][v] - mean[v]) * inv[v] * g + b;
        x += P.actors[(size_t)(rw + t * 16 + v + 8 * hi) * D + col];   // residual (f32, L2)
        sX[(wr + t * 16 + v + 8 * hi) * SA + col] = f2bf(fmaxf(x, 0.f));
      }
    }
  }
  __syncthreads();

  // ---- stage 3: preds = h @ Wo[m] + bo ; reg = preds + ctr -> reg_ws ; dist = -(pred+bo)
  loadW(sW, P.pWo + (size_t)mode * D * 60, D, 60, tid);
  for (int idx = tid; idx < 4 * D; idx += TPB) {                 // zero-pad cols 60..63
    int n = 60 + (idx >> 7), k = idx & 127;
    sW[n * SA + k] = 0;
  }
  loadV(sC, P.pbo + mode * 60, 60, tid);
  __syncthreads();
  prefW(P.adW2, D * D, tid);
#pragma unroll
  for (int nt = 0; nt < 4; ++nt) { acc0[nt] = zf; acc1[nt] = zf; }
  {
    const unsigned short* r0p = sX + (wr + lo) * SA;
    const unsigned short* r1p = sX + (wr + 16 + lo) * SA;
#pragma unroll
    for (int kt = 0; kt < 4; ++kt) {
      bf16x16 fa0 = fragA(r0p, hi, kt);
      bf16x16 fa1 = fragA(r1p, hi, kt);
#pragma unroll
      for (int nt = 0; nt < 4; ++nt) {
        bf16x16 b = fragB(sW + (nt * 16 + lo) * SA, hi, kt);
        acc0[nt] = __builtin_amdgcn_wmma_f32_16x16x32_bf16(
            false, fa0, false, b, (short)0, acc0[nt], false, false);
        acc1[nt] = __builtin_amdgcn_wmma_f32_16x16x32_bf16(
            false, fa1, false, b, (short)0, acc1[nt], false, false);
      }
    }
  }
#pragma unroll
  for (int t = 0; t < 2; ++t) {
    f32x8* A = t ? acc1 : acc0;
#pragma unroll
    for (int nt = 0; nt < 4; ++nt) {
      int o = nt * 16 + lo;
      if (o < 60) {
        float bo = sC[o];
#pragma unroll
        for (int v = 0; v < 8; ++v) {
          int gr = rw + t * 16 + v + 8 * hi;
          reg_ws[(size_t)gr * (MODES * 60) + mode * 60 + o] =
              A[nt][v] + bo + P.ctrs[(size_t)gr * 2 + (o & 1)];
        }
      }
    }
    if (lo == 10) { float bo = sC[58];
#pragma unroll
      for (int v = 0; v < 8; ++v) sDist[wr + t * 16 + v + 8 * hi][0] = -(A[3][v] + bo); }
    if (lo == 11) { float bo = sC[59];
#pragma unroll
      for (int v = 0; v < 8; ++v) sDist[wr + t * 16 + v + 8 * hi][1] = -(A[3][v] + bo); }
  }
  __syncthreads();

  // ---- stage 4: dist = relu(dist2 @ dW1 + db1) -> sX   (K=2: plain VALU)
  loadV(sG, P.adW1,     D, tid);     // dW1 row k=0
  loadV(sB, P.adW1 + D, D, tid);     // dW1 row k=1
  loadV(sC, P.adb1,     D, tid);
  __syncthreads();
  for (int e = lane; e < 32 * D; e += 32) {
    int r = e >> 7, k = e & 127;
    float dx = sDist[wr + r][0], dy = sDist[wr + r][1];
    sX[(wr + r) * SA + k] = f2bf(fmaxf(dx * sG[k] + dy * sB[k] + sC[k], 0.f));
  }
  __syncthreads();

  // ---- stage 5: dist = relu(GN(dist @ dW2, dg2, db2)) -> sX
  loadW(sW, P.adW2, D, D, tid);
  loadV(sG, P.adg2, D, tid);
  loadV(sB, P.adb2, D, tid);
  __syncthreads();
  prefW(P.aaW, 2 * D * D, tid);
#pragma unroll
  for (int nt = 0; nt < 8; ++nt) { acc0[nt] = zf; acc1[nt] = zf; }
  mm128x2(acc0, acc1, sX + (wr + lo) * SA, sX + (wr + 16 + lo) * SA, sW, lo, hi);
  gn_relu_store(acc0, sG, sB, sX + wr * SA, lo, hi);
  gn_relu_store(acc1, sG, sB, sX + (wr + 16) * SA, lo, hi);
  __syncthreads();

  // ---- stage 6: feats = relu(GN(concat(dist, actors) @ aW, ag, ab)) -> sX
  loadW(sW, P.aaW, D, D, tid);                      // aW rows k=0..127 (dist half)
  __syncthreads();
#pragma unroll
  for (int nt = 0; nt < 8; ++nt) { acc0[nt] = zf; acc1[nt] = zf; }
  mm128x2(acc0, acc1, sX + (wr + lo) * SA, sX + (wr + 16 + lo) * SA, sW, lo, hi);
  __syncthreads();
  loadW(sW, P.aaW + (size_t)D * D, D, D, tid);      // aW rows k=128..255 (actor half)
  loadV(sG, P.aag, D, tid);
  loadV(sB, P.aab, D, tid);
  __syncthreads();
  prefW(P.cW1, D * D, tid);
  mm128x2(acc0, acc1, sAct + (wr + lo) * SA, sAct + (wr + 16 + lo) * SA, sW, lo, hi);
  gn_relu_store(acc0, sG, sB, sX + wr * SA, lo, hi);
  gn_relu_store(acc1, sG, sB, sX + (wr + 16) * SA, lo, hi);
  __syncthreads();

  // ---- stage 7: c = relu(GN(feats @ cls_W1, g1, b1)) -> sAct (actors dead now)
  loadW(sW, P.cW1, D, D, tid);
  loadV(sG, P.cg1, D, tid);
  loadV(sB, P.cb1, D, tid);
  __syncthreads();
  prefW(P.cW2, D * D, tid);
#pragma unroll
  for (int nt = 0; nt < 8; ++nt) { acc0[nt] = zf; acc1[nt] = zf; }
  mm128x2(acc0, acc1, sX + (wr + lo) * SA, sX + (wr + 16 + lo) * SA, sW, lo, hi);
  gn_relu_store(acc0, sG, sB, sAct + wr * SA, lo, hi);
  gn_relu_store(acc1, sG, sB, sAct + (wr + 16) * SA, lo, hi);
  __syncthreads();

  // ---- stage 8: c = relu(GN(c @ cls_W2, g2, b2) + feats)   (kept in acc)
  loadW(sW, P.cW2, D, D, tid);
  loadV(sG, P.cg2, D, tid);
  loadV(sB, P.cb2, D, tid);
  loadV(sC, P.cWo, D, tid);
  __syncthreads();
#pragma unroll
  for (int nt = 0; nt < 8; ++nt) { acc0[nt] = zf; acc1[nt] = zf; }
  mm128x2(acc0, acc1, sAct + (wr + lo) * SA, sAct + (wr + 16 + lo) * SA, sW, lo, hi);
#pragma unroll
  for (int t = 0; t < 2; ++t) {
    f32x8* A = t ? acc1 : acc0;
    gn_stats(A, mean, inv);
#pragma unroll
    for (int nt = 0; nt < 8; ++nt) {
      int col = nt * 16 + lo; float g = sG[col], b = sB[col];
#pragma unroll
      for (int v = 0; v < 8; ++v) {
        float x = (A[nt][v] - mean[v]) * inv[v] * g + b;
        x += bf2f(sX[(wr + t * 16 + v + 8 * hi) * SA + col]);   // + feats
        A[nt][v] = fmaxf(x, 0.f);
      }
    }
  }

  // ---- stage 9: cls = c @ cls_Wo + cls_bo -> cls_ws[n][mode]
  float cbo = P.cbo[0];
#pragma unroll
  for (int t = 0; t < 2; ++t) {
    f32x8* A = t ? acc1 : acc0;
#pragma unroll
    for (int v = 0; v < 8; ++v) {
      float p = 0.f;
#pragma unroll
      for (int nt = 0; nt < 8; ++nt) p += A[nt][v] * sC[nt * 16 + lo];
#pragma unroll
      for (int m = 1; m < 16; m <<= 1) p += __shfl_xor(p, m, 32);
      if (lo == 0) cls_ws[(size_t)(rw + t * 16 + v + 8 * hi) * MODES + mode] = p + cbo;
    }
  }
}

// ---------------------------------------------------------------------------
// Kernel 2: stable descending 6-way argsort per actor; emit cls + permuted reg.
// ---------------------------------------------------------------------------
__global__ __launch_bounds__(TPB) void lanegcn_sort_kernel(const float* __restrict__ cls_ws,
                                                           const float* __restrict__ reg_ws,
                                                           float* __restrict__ out_cls,
                                                           float* __restrict__ out_reg) {
  __shared__ int sOrd[ACT_BLK][MODES];
  const int tid  = threadIdx.x;
  const int base = blockIdx.x * ACT_BLK;
  if (tid < ACT_BLK) {
    int a = base + tid;
    float key[MODES]; int ord[MODES];
#pragma unroll
    for (int j = 0; j < MODES; ++j) { key[j] = cls_ws[(size_t)a * MODES + j]; ord[j] = j; }
#pragma unroll
    for (int i = 1; i < MODES; ++i) {                 // stable insertion sort, descending
      float kv = key[i]; int ov = ord[i]; int j = i - 1;
      while (j >= 0 && key[j] < kv) { key[j + 1] = key[j]; ord[j + 1] = ord[j]; --j; }
      key[j + 1] = kv; ord[j + 1] = ov;
    }
#pragma unroll
    for (int j = 0; j < MODES; ++j) {
      out_cls[(size_t)a * MODES + j] = key[j];
      sOrd[tid][j] = ord[j];
    }
  }
  __syncthreads();
  for (int idx = tid; idx < ACT_BLK * MODES * 60; idx += TPB) {
    int la  = idx / (MODES * 60);
    int rem = idx - la * (MODES * 60);
    int j   = rem / 60, k = rem - j * 60;
    int a   = base + la;
    out_reg[(size_t)a * (MODES * 60) + j * 60 + k] =
        reg_ws[(size_t)a * (MODES * 60) + sOrd[la][j] * 60 + k];
  }
}

extern "C" void kernel_launch(void* const* d_in, const int* in_sizes, int n_in,
                              void* d_out, int out_size, void* d_ws, size_t ws_size,
                              hipStream_t stream) {
  (void)in_sizes; (void)n_in; (void)out_size; (void)ws_size;
  HeadParams P;
  P.actors = (const float*)d_in[0];
  P.ctrs   = (const float*)d_in[1];
  // d_in[2] = actor_idcs: identity partition of arange(N) -> no-op in the math
  P.pW1 = (const float*)d_in[3];  P.pg1 = (const float*)d_in[4];  P.pb1 = (const float*)d_in[5];
  P.pW2 = (const float*)d_in[6];  P.pg2 = (const float*)d_in[7];  P.pb2 = (const float*)d_in[8];
  P.pWo = (const float*)d_in[9];  P.pbo = (const float*)d_in[10];
  P.cW1 = (const float*)d_in[11]; P.cg1 = (const float*)d_in[12]; P.cb1 = (const float*)d_in[13];
  P.cW2 = (const float*)d_in[14]; P.cg2 = (const float*)d_in[15]; P.cb2 = (const float*)d_in[16];
  P.cWo = (const float*)d_in[17]; P.cbo = (const float*)d_in[18];
  P.adW1 = (const float*)d_in[19]; P.adb1 = (const float*)d_in[20];
  P.adW2 = (const float*)d_in[21]; P.adg2 = (const float*)d_in[22]; P.adb2 = (const float*)d_in[23];
  P.aaW  = (const float*)d_in[24]; P.aag  = (const float*)d_in[25]; P.aab  = (const float*)d_in[26];

  float* reg_ws = (float*)d_ws;                                  // [N][6*60]
  float* cls_ws = reg_ws + (size_t)NA * MODES * 60;              // [N][6]

  dim3 g1(NA / ROWS_WG, MODES);
  lanegcn_head_kernel<<<g1, TPB, 0, stream>>>(P, reg_ws, cls_ws);

  float* out_cls = (float*)d_out;                                // [N][6]
  float* out_reg = out_cls + (size_t)NA * MODES;                 // [N][6][30][2]
  lanegcn_sort_kernel<<<NA / ACT_BLK, TPB, 0, stream>>>(cls_ws, reg_ws, out_cls, out_reg);
}
